// TransformerBlockV2_90056874262689
// MI455X (gfx1250) — matrix operand order repaired and checked by reading, not compiled, not run on hardware
//
#include <hip/hip_runtime.h>

// ---------------------------------------------------------------------------
// TransformerBlock for MI455X (gfx1250): all GEMMs on v_wmma_f32_16x16x32_bf16
// B=2, S=2048, E=2048, H=16, D=128, I=8192
// ---------------------------------------------------------------------------

typedef __attribute__((ext_vector_type(16))) __bf16 v16bf;
typedef __attribute__((ext_vector_type(8)))  float  v8f;

union BFrag { v16bf v; uint4 q[2]; };

__device__ __forceinline__ unsigned short f2bf(float f) {
    // round-to-nearest-even f32 -> bf16
    unsigned int u = __float_as_uint(f);
    return (unsigned short)((u + 0x7FFFu + ((u >> 16) & 1u)) >> 16);
}

__device__ __forceinline__ v8f wmma_bf16(const BFrag& a, const BFrag& b, v8f c) {
    return __builtin_amdgcn_wmma_f32_16x16x32_bf16(false, a.v, false, b.v,
                                                   (short)0, c, false, false);
}

// ---------------------------------------------------------------------------
// Weight convert + transpose: in[K][N] f32 -> out[N][K] bf16
// grid = (N/256, K); coalesced reads, strided bf16 writes (one pass, L2-bound)
// ---------------------------------------------------------------------------
__global__ __launch_bounds__(256) void cvt_transpose(const float* __restrict__ in,
                                                     unsigned short* __restrict__ out,
                                                     int K, int N) {
    int n = blockIdx.x * 256 + threadIdx.x;
    int k = blockIdx.y;
    out[(size_t)n * K + k] = f2bf(in[(size_t)k * N + n]);
}

// ---------------------------------------------------------------------------
// LayerNorm over E=2048, one block per row, f32 in -> bf16 out
// ---------------------------------------------------------------------------
__global__ __launch_bounds__(256) void layernorm_bf16(const float* __restrict__ x,
                                                      const float* __restrict__ w,
                                                      const float* __restrict__ bia,
                                                      unsigned short* __restrict__ out) {
    const int row = blockIdx.x;
    const float* xr = x + (size_t)row * 2048;
    float vals[8];
    float sum = 0.f, sq = 0.f;
    for (int i = 0; i < 8; ++i) {
        float v = xr[threadIdx.x + i * 256];
        vals[i] = v; sum += v; sq += v * v;
    }
    __shared__ float red[256], red2[256];
    red[threadIdx.x] = sum; red2[threadIdx.x] = sq;
    __syncthreads();
    for (int s = 128; s > 0; s >>= 1) {
        if ((int)threadIdx.x < s) {
            red[threadIdx.x]  += red[threadIdx.x + s];
            red2[threadIdx.x] += red2[threadIdx.x + s];
        }
        __syncthreads();
    }
    float mu  = red[0] * (1.0f / 2048.0f);
    float var = red2[0] * (1.0f / 2048.0f) - mu * mu;
    float inv = rsqrtf(var + 1e-5f);
    for (int i = 0; i < 8; ++i) {
        int c = threadIdx.x + i * 256;
        out[(size_t)row * 2048 + c] = f2bf((vals[i] - mu) * inv * w[c] + bia[c]);
    }
}

// ---------------------------------------------------------------------------
// WMMA GEMM: C[M,N] = A[M,K](bf16, row-major) x BT[N,K](bf16, i.e. B^T)
// Block = 256 threads (8 waves, 2x4), 128x256 tile; wave = 64(M) x 64(N):
// 16 WMMAs per 8 fragment loads per K=32 step.
// Epilogues: mode 0 = QKV scatter, mode 1 = +resid -> f32, mode 2 = GELU -> bf16
// ---------------------------------------------------------------------------
__global__ __launch_bounds__(256) void gemm_bf16_wmma(
    const unsigned short* __restrict__ A, const unsigned short* __restrict__ BT,
    int N, int K, int mode,
    const float* __restrict__ resid, float* __restrict__ outF,
    unsigned short* __restrict__ outH,
    unsigned short* __restrict__ qO, unsigned short* __restrict__ kO,
    unsigned short* __restrict__ vTO, float qscale) {

    const int lane = threadIdx.x & 31;
    const int w    = threadIdx.x >> 5;
    const int hh   = lane >> 4;   // lane half
    const int ln   = lane & 15;   // m (A frags) / n (B,C frags)
    const int wm   = w >> 2, wn = w & 3;
    const int row0 = blockIdx.y * 128 + wm * 64;
    const int col0 = blockIdx.x * 256 + wn * 64;

    v8f acc[4][4];
    for (int i = 0; i < 4; ++i)
        for (int j = 0; j < 4; ++j) acc[i][j] = {};

    for (int kk = 0; kk < K; kk += 32) {
        BFrag af[4], bf[4];
        // A fragments (contiguous K = 16h..16h+15, matching B)
        for (int i = 0; i < 4; ++i) {
            const unsigned short* p = A + (size_t)(row0 + i * 16 + ln) * K + kk;
            af[i].q[0] = *(const uint4*)(p + 16 * hh);
            af[i].q[1] = *(const uint4*)(p + 16 * hh + 8);
        }
        // B fragments from B^T: lane needs contiguous K = 16h..16h+15
        for (int j = 0; j < 4; ++j) {
            const unsigned short* p = BT + (size_t)(col0 + j * 16 + ln) * K + kk + 16 * hh;
            bf[j].q[0] = *(const uint4*)(p);
            bf[j].q[1] = *(const uint4*)(p + 8);
        }
        for (int i = 0; i < 4; ++i)
            for (int j = 0; j < 4; ++j)
                acc[i][j] = wmma_bf16(af[i], bf[j], acc[i][j]);
    }

    // Epilogue. C layout: VGPR r, lane -> row = r + 8*hh, col = ln (per subtile)
    for (int i = 0; i < 4; ++i) {
        for (int j = 0; j < 4; ++j) {
            for (int r = 0; r < 8; ++r) {
                int row = row0 + i * 16 + 8 * hh + r;
                int col = col0 + j * 16 + ln;
                float v = acc[i][j][r];
                if (mode == 0) {
                    // col in [0, 6144): (which, head, d); row in [0,4096): (b, s)
                    int which = col >> 11;
                    int hd    = col & 2047;
                    int head  = hd >> 7;
                    int d     = hd & 127;
                    int bb    = row >> 11;
                    int s     = row & 2047;
                    size_t o = (((size_t)bb * 16 + head) * 2048 + s) * 128 + d;
                    if (which == 0)      qO[o] = f2bf(v * qscale);
                    else if (which == 1) kO[o] = f2bf(v);
                    else vTO[(((size_t)bb * 16 + head) * 128 + d) * 2048 + s] = f2bf(v);
                } else if (mode == 1) {
                    size_t idx = (size_t)row * N + col;
                    outF[idx] = resid[idx] + v;
                } else {
                    size_t idx = (size_t)row * N + col;
                    // gelu(tanh approx); tanh(u) = 1 - 2/(exp(2u)+1) (saturates right)
                    float u = 0.7978845608028654f * (v + 0.044715f * v * v * v);
                    float e = __expf(2.0f * u);
                    float t = 1.0f - 2.0f / (e + 1.0f);
                    outH[idx] = f2bf(0.5f * v * (1.0f + t));
                }
            }
        }
    }
}

// ---------------------------------------------------------------------------
// Flash attention: one wave per (b, head, 16-query tile). Causal.
// Q,K: bf16 [B,H,S,D] (Q pre-scaled by D^-0.5). Vt: bf16 [B,H,D,S].
// O: bf16 [B,S,H,D].
// ---------------------------------------------------------------------------
__global__ __launch_bounds__(32) void attn_flash_wmma(
    const unsigned short* __restrict__ Q, const unsigned short* __restrict__ Kc,
    const unsigned short* __restrict__ Vt, unsigned short* __restrict__ O) {

    const int lane = threadIdx.x;
    const int hh = lane >> 4;
    const int ln = lane & 15;
    const int qb = blockIdx.x, head = blockIdx.y, b = blockIdx.z;
    const size_t bh = (size_t)(b * 16 + head);
    const unsigned short* Qbase = Q  + (bh * 2048 + (size_t)qb * 16) * 128;
    const unsigned short* Kbase = Kc + bh * 2048 * 128;
    const unsigned short* Vbase = Vt + bh * 128 * 2048;

    // Q fragments: 16 rows x 128 (4 chunks of K=32)
    BFrag qf[4];
    for (int c = 0; c < 4; ++c) {
        const unsigned short* p = Qbase + (size_t)ln * 128 + 32 * c;
        qf[c].q[0] = *(const uint4*)(p + 16 * hh);
        qf[c].q[1] = *(const uint4*)(p + 16 * hh + 8);
    }

    v8f acco[8];
    for (int i = 0; i < 8; ++i) acco[i] = {};
    float m_r[8], l_r[8];
    for (int r = 0; r < 8; ++r) { m_r[r] = -__builtin_inff(); l_r[r] = 0.f; }

    __shared__ __align__(16) unsigned short Plds[16 * 32];

    const int row_hi = qb * 16 + 15;
    const int ntiles = row_hi / 32 + 1;     // 32 keys per tile, causal bound

    for (int t = 0; t < ntiles; ++t) {
        // scores: 16 queries x 32 keys = 2 subtiles, 4 WMMAs each (D=128)
        v8f sc[2]; sc[0] = {}; sc[1] = {};
        for (int sub = 0; sub < 2; ++sub) {
            const unsigned short* kp = Kbase + (size_t)(t * 32 + sub * 16 + ln) * 128;
            for (int c = 0; c < 4; ++c) {
                BFrag bk;
                bk.q[0] = *(const uint4*)(kp + 32 * c + 16 * hh);
                bk.q[1] = *(const uint4*)(kp + 32 * c + 16 * hh + 8);
                sc[sub] = wmma_bf16(qf[c], bk, sc[sub]);
            }
        }
        // causal mask + online softmax (rows live on 16-lane halves)
        for (int r = 0; r < 8; ++r) {
            int row = qb * 16 + 8 * hh + r;
            float s0 = sc[0][r], s1 = sc[1][r];
            int c0 = t * 32 + ln;
            if (c0 > row)      s0 = -__builtin_inff();
            if (c0 + 16 > row) s1 = -__builtin_inff();
            float mx = fmaxf(s0, s1);
            for (int off = 1; off < 16; off <<= 1)
                mx = fmaxf(mx, __shfl_xor(mx, off, 32));
            float mnew = fmaxf(m_r[r], mx);
            float corr = __expf(m_r[r] - mnew);
            float p0 = __expf(s0 - mnew);
            float p1 = __expf(s1 - mnew);
            float rs = p0 + p1;
            for (int off = 1; off < 16; off <<= 1)
                rs += __shfl_xor(rs, off, 32);
            l_r[r] = l_r[r] * corr + rs;
            m_r[r] = mnew;
            for (int dn = 0; dn < 8; ++dn) acco[dn][r] *= corr;
            Plds[(8 * hh + r) * 32 + ln]      = f2bf(p0);
            Plds[(8 * hh + r) * 32 + 16 + ln] = f2bf(p1);
        }
        __syncthreads();
        // re-fragment P (C layout -> A layout) via LDS
        BFrag pf;
        {
            const unsigned short* pp = Plds + ln * 32;
            pf.q[0] = *(const uint4*)(pp + 16 * hh);
            pf.q[1] = *(const uint4*)(pp + 16 * hh + 8);
        }
        // P(16x32) x V(32x128): 8 WMMAs against V^T (contiguous key fragments)
        for (int dn = 0; dn < 8; ++dn) {
            const unsigned short* vp = Vbase + (size_t)(dn * 16 + ln) * 2048 + t * 32 + 16 * hh;
            BFrag vf;
            vf.q[0] = *(const uint4*)(vp);
            vf.q[1] = *(const uint4*)(vp + 8);
            acco[dn] = wmma_bf16(pf, vf, acco[dn]);
        }
        __syncthreads();
    }

    float linv[8];
    for (int r = 0; r < 8; ++r) linv[r] = 1.0f / l_r[r];
    for (int dn = 0; dn < 8; ++dn) {
        for (int r = 0; r < 8; ++r) {
            int s = qb * 16 + 8 * hh + r;
            float val = acco[dn][r] * linv[r];
            // O layout [B,S,H,D] == [B,S,E] row-major
            O[(((size_t)b * 2048 + s) * 16 + head) * 128 + dn * 16 + ln] = f2bf(val);
        }
    }
}

// ---------------------------------------------------------------------------
extern "C" void kernel_launch(void* const* d_in, const int* in_sizes, int n_in,
                              void* d_out, int out_size, void* d_ws, size_t ws_size,
                              hipStream_t stream) {
    (void)in_sizes; (void)n_in; (void)out_size; (void)ws_size;
    const float* x     = (const float*)d_in[0];
    const float* ln1w  = (const float*)d_in[1];
    const float* ln1b  = (const float*)d_in[2];
    const float* Wqkv  = (const float*)d_in[3];
    const float* Wo    = (const float*)d_in[4];
    const float* ln2w  = (const float*)d_in[5];
    const float* ln2b  = (const float*)d_in[6];
    const float* W1    = (const float*)d_in[7];
    const float* W2    = (const float*)d_in[8];

    const int M = 4096;      // B*S
    const int E = 2048, I = 8192, N3 = 6144;

    size_t off = 0;
    auto alloc = [&](size_t bytes) -> char* {
        char* p = (char*)d_ws + off;
        off += (bytes + 255) & ~(size_t)255;
        return p;
    };
    unsigned short* hA    = (unsigned short*)alloc((size_t)M * E * 2);
    unsigned short* WqkvT = (unsigned short*)alloc((size_t)N3 * E * 2);
    unsigned short* WoT   = (unsigned short*)alloc((size_t)E * E * 2);
    unsigned short* W1T   = (unsigned short*)alloc((size_t)I * E * 2);
    unsigned short* W2T   = (unsigned short*)alloc((size_t)E * I * 2);
    unsigned short* Qb    = (unsigned short*)alloc((size_t)M * E * 2);
    unsigned short* Kb    = (unsigned short*)alloc((size_t)M * E * 2);
    unsigned short* VTb   = (unsigned short*)alloc((size_t)M * E * 2);
    unsigned short* Ob    = (unsigned short*)alloc((size_t)M * E * 2);
    float*          x1    = (float*)alloc((size_t)M * E * 4);
    unsigned short* h2    = (unsigned short*)alloc((size_t)M * E * 2);
    unsigned short* g     = (unsigned short*)alloc((size_t)M * I * 2);

    // 1) weights -> bf16, transposed to [N][K]
    cvt_transpose<<<dim3(N3 / 256, E), 256, 0, stream>>>(Wqkv, WqkvT, E, N3);
    cvt_transpose<<<dim3(E  / 256, E), 256, 0, stream>>>(Wo,   WoT,   E, E);
    cvt_transpose<<<dim3(I  / 256, E), 256, 0, stream>>>(W1,   W1T,   E, I);
    cvt_transpose<<<dim3(E  / 256, I), 256, 0, stream>>>(W2,   W2T,   I, E);

    // 2) LN1
    layernorm_bf16<<<M, 256, 0, stream>>>(x, ln1w, ln1b, hA);

    // 3) QKV projection (scatter epilogue, Q pre-scaled by D^-0.5)
    gemm_bf16_wmma<<<dim3(N3 / 256, M / 128), 256, 0, stream>>>(
        hA, WqkvT, N3, E, 0, nullptr, nullptr, nullptr,
        Qb, Kb, VTb, 0.08838834764831845f /* 128^-0.5 */);

    // 4) flash attention
    attn_flash_wmma<<<dim3(128, 16, 2), 32, 0, stream>>>(Qb, Kb, VTb, Ob);

    // 5) output projection + residual -> x1 (f32)
    gemm_bf16_wmma<<<dim3(E / 256, M / 128), 256, 0, stream>>>(
        Ob, WoT, E, E, 1, x, x1, nullptr, nullptr, nullptr, nullptr, 0.f);

    // 6) LN2
    layernorm_bf16<<<M, 256, 0, stream>>>(x1, ln2w, ln2b, h2);

    // 7) FFN up + GELU -> g (bf16)
    gemm_bf16_wmma<<<dim3(I / 256, M / 128), 256, 0, stream>>>(
        h2, W1T, I, E, 2, nullptr, nullptr, g, nullptr, nullptr, nullptr, 0.f);

    // 8) FFN down + residual -> d_out (f32)
    gemm_bf16_wmma<<<dim3(E / 256, M / 128), 256, 0, stream>>>(
        g, W2T, E, I, 1, x1, (float*)d_out, nullptr, nullptr, nullptr, nullptr, 0.f);
}
